// RandomVanillaAttention_16286515986791
// MI455X (gfx1250) — matrix operand order repaired and checked by reading, not compiled, not run on hardware
//
#include <hip/hip_runtime.h>

// Problem constants
static constexpr int Bc = 16;
static constexpr int Sc = 2048;
static constexpr int Dc = 512;
static constexpr int LW = 40;  // LDS row stride in halves (80B, 16B-aligned)

typedef __attribute__((ext_vector_type(16))) __bf16     v16bf;
typedef __attribute__((ext_vector_type(8)))  float      v8f;
typedef __attribute__((ext_vector_type(8)))  unsigned   v8u;

__device__ __forceinline__ unsigned short f2bf(float f) {
    unsigned u = __builtin_bit_cast(unsigned, f);
    u += 0x7FFFu + ((u >> 16) & 1u);          // round-to-nearest-even
    return (unsigned short)(u >> 16);
}
__device__ __forceinline__ unsigned pack2(float lo, float hi) {
    return (unsigned)f2bf(lo) | ((unsigned)f2bf(hi) << 16);
}

__device__ __forceinline__ v8f wmma_bf16(v16bf a, v16bf b, v8f c) {
    return __builtin_amdgcn_wmma_f32_16x16x32_bf16(false, a, false, b, (short)0, c, false, false);
}

// ---- CDNA5 async global->LDS copy (ASYNCcnt path) -------------------------
// Generic LDS pointers carry the LDS byte offset in addr[31:0] (aperture map),
// which is what the instruction's VDST (LDS address) operand expects.
__device__ __forceinline__ void async_copy_b128(const unsigned short* lds,
                                                const unsigned short* gsrc) {
    unsigned ldsOff = (unsigned)(size_t)lds;
    unsigned long long ga = (unsigned long long)(size_t)gsrc;
    asm volatile("global_load_async_to_lds_b128 %0, %1, off"
                 :: "v"(ldsOff), "v"(ga) : "memory");
}
__device__ __forceinline__ void wait_async() {
    asm volatile("s_wait_asynccnt 0x0" ::: "memory");
}

// A-matrix 16x32 bf16 fragment: lane M=L%16, h=L/16; VGPR v: K=(v/4)*16+h*8+(v%4)*2
__device__ __forceinline__ v16bf load_frag_A(const unsigned short* lds, int row, int lane) {
    int h = lane >> 4;
    uint4 a = *(const uint4*)(lds + row * LW + h * 8);
    uint4 b = *(const uint4*)(lds + row * LW + 16 + h * 8);
    v8u u; u[0]=a.x; u[1]=a.y; u[2]=a.z; u[3]=a.w; u[4]=b.x; u[5]=b.y; u[6]=b.z; u[7]=b.w;
    return __builtin_bit_cast(v16bf, u);
}
// B-matrix 32x16 bf16 fragment: lane N=L%16, h=L/16; VGPR v: K=(v/4)*16+h*8+(v%4)*2 (row-major [n][k] tile)
__device__ __forceinline__ v16bf load_frag_B(const unsigned short* lds, int row, int lane) {
    int h = lane >> 4;
    uint4 a = *(const uint4*)(lds + row * LW + h * 8);
    uint4 b = *(const uint4*)(lds + row * LW + 16 + h * 8);
    v8u u; u[0]=a.x; u[1]=a.y; u[2]=a.z; u[3]=a.w; u[4]=b.x; u[5]=b.y; u[6]=b.z; u[7]=b.w;
    return __builtin_bit_cast(v16bf, u);
}

// Stage ROWSx32 fp32 tile -> bf16 LDS tile (VALU convert + ds_store, constant trip count)
template <int ROWS, int NT>
__device__ __forceinline__ void stage_f32(unsigned short* lds, const float* src,
                                          int srcStride, int tid) {
    constexpr int ITER = ROWS * 32 / 4 / NT;
#pragma unroll
    for (int j = 0; j < ITER; ++j) {
        int i = j * NT + tid;
        int row = i >> 3, c4 = i & 7;
        float4 v = *(const float4*)(src + (size_t)row * srcStride + c4 * 4);
        uint2 w; w.x = pack2(v.x, v.y); w.y = pack2(v.z, v.w);
        *(uint2*)(lds + row * LW + c4 * 4) = w;
    }
}
// Stage ROWSx32 bf16 tile -> LDS via async copies (16B per lane)
template <int ROWS, int NT>
__device__ __forceinline__ void stage_async(unsigned short* lds, const unsigned short* src,
                                            int srcStride, int tid) {
    constexpr int ITER = ROWS * 32 / 8 / NT;
#pragma unroll
    for (int j = 0; j < ITER; ++j) {
        int i = j * NT + tid;
        int row = i >> 2, c8 = i & 3;
        async_copy_b128(lds + row * LW + c8 * 8, src + (size_t)row * srcStride + c8 * 8);
    }
}

// K0: value fp32 [b][k][d] -> vbT bf16 [b][d][k]  (LDS-tiled transpose + convert)
__global__ __launch_bounds__(256) void transpose_v(const float* __restrict__ value,
                                                   unsigned short* __restrict__ vbT) {
    __shared__ __align__(16) float tile[64 * 68];
    int b = blockIdx.z, k0 = blockIdx.x * 64, d0 = blockIdx.y * 64;
    const float* src = value + (size_t)b * Sc * Dc;
    unsigned short* dst = vbT + (size_t)b * Sc * Dc;
    int tid = threadIdx.x;
#pragma unroll
    for (int i = 0; i < 4; ++i) {
        int idx = i * 256 + tid;
        int r = idx >> 4, c4 = idx & 15;
        float4 v = *(const float4*)(src + (size_t)(k0 + r) * Dc + d0 + c4 * 4);
        *(float4*)(tile + r * 68 + c4 * 4) = v;
    }
    __syncthreads();
#pragma unroll
    for (int i = 0; i < 4; ++i) {
        int idx = i * 256 + tid;
        int r = idx >> 4, c4 = idx & 15;     // r = d index, c4*4 = k chunk
        float x0 = tile[(c4 * 4 + 0) * 68 + r];
        float x1 = tile[(c4 * 4 + 1) * 68 + r];
        float x2 = tile[(c4 * 4 + 2) * 68 + r];
        float x3 = tile[(c4 * 4 + 3) * 68 + r];
        uint2 w; w.x = pack2(x0, x1); w.y = pack2(x2, x3);
        *(uint2*)(dst + (size_t)(d0 + r) * Sc + k0 + c4 * 4) = w;
    }
}

// K1: projections  out[s][e] = X[s][:] . W[e][:] + bias[e]   (64x64 tiles, 4 waves)
__global__ __launch_bounds__(128) void proj(const float* __restrict__ query,
                                            const float* __restrict__ key,
                                            const float* __restrict__ Wq, const float* __restrict__ bq,
                                            const float* __restrict__ Wk, const float* __restrict__ bk,
                                            unsigned short* __restrict__ qb,
                                            unsigned short* __restrict__ kb) {
    __shared__ __align__(16) unsigned short ldsA[64 * LW];
    __shared__ __align__(16) unsigned short ldsB[64 * LW];
    int z = blockIdx.z, b = z & 15;
    bool isK = z >= Bc;
    const float* X = (isK ? key : query) + (size_t)b * Sc * Dc;
    const float* W = isK ? Wk : Wq;
    const float* bias = isK ? bk : bq;
    unsigned short* out = (isK ? kb : qb) + (size_t)b * Sc * Dc;
    int row0 = blockIdx.x * 64, col0 = blockIdx.y * 64;
    int tid = threadIdx.x, lane = tid & 31, wave = tid >> 5;
    v8f acc[4] = {};
    for (int k0 = 0; k0 < Dc; k0 += 32) {
        __syncthreads();
        stage_f32<64, 128>(ldsA, X + (size_t)row0 * Dc + k0, Dc, tid);
        stage_f32<64, 128>(ldsB, W + (size_t)col0 * Dc + k0, Dc, tid);
        __syncthreads();
        v16bf a = load_frag_A(ldsA, wave * 16 + (lane & 15), lane);
        v16bf bf[4];
#pragma unroll
        for (int t = 0; t < 4; ++t) bf[t] = load_frag_B(ldsB, t * 16 + (lane & 15), lane);
#pragma unroll
        for (int t = 0; t < 4; ++t) acc[t] = wmma_bf16(a, bf[t], acc[t]);
    }
    int h = lane >> 4, n = lane & 15;
#pragma unroll
    for (int t = 0; t < 4; ++t) {
        int col = col0 + t * 16 + n;
        float bv = bias[col];
#pragma unroll
        for (int v = 0; v < 8; ++v) {
            int row = row0 + wave * 16 + h * 8 + v;
            out[(size_t)row * Dc + col] = f2bf(acc[t][v] + bv);
        }
    }
}

// K2: scores = attn_bias + Q.K^T
// 128x128 tiles, 8 waves, double-buffered LDS fed by async global->LDS copies.
__global__ __launch_bounds__(256) void scores(const unsigned short* __restrict__ qb,
                                              const unsigned short* __restrict__ kb,
                                              const float* __restrict__ attn_bias,
                                              float* __restrict__ attn) {
    __shared__ __align__(16) unsigned short ldsQ[2][128 * LW];
    __shared__ __align__(16) unsigned short ldsK[2][128 * LW];
    int b = blockIdx.z, q0 = blockIdx.x * 128, k0 = blockIdx.y * 128;
    const unsigned short* Q = qb + (size_t)b * Sc * Dc;
    const unsigned short* K = kb + (size_t)b * Sc * Dc;
    float* out = attn + (size_t)b * Sc * Sc;
    int tid = threadIdx.x, lane = tid & 31, wave = tid >> 5;
    v8f acc[8] = {};
    constexpr int NIT = Dc / 32;
    stage_async<128, 256>(ldsQ[0], Q + (size_t)q0 * Dc, Dc, tid);
    stage_async<128, 256>(ldsK[0], K + (size_t)k0 * Dc, Dc, tid);
    wait_async();
    __syncthreads();
    for (int it = 0; it < NIT; ++it) {
        int cur = it & 1;
        if (it + 1 < NIT) {   // prefetch next K-tile into the other buffer (async)
            stage_async<128, 256>(ldsQ[cur ^ 1], Q + (size_t)q0 * Dc + (it + 1) * 32, Dc, tid);
            stage_async<128, 256>(ldsK[cur ^ 1], K + (size_t)k0 * Dc + (it + 1) * 32, Dc, tid);
        }
        v16bf a = load_frag_A(ldsQ[cur], wave * 16 + (lane & 15), lane);
        v16bf bf[8];
#pragma unroll
        for (int t = 0; t < 8; ++t) bf[t] = load_frag_B(ldsK[cur], t * 16 + (lane & 15), lane);
#pragma unroll
        for (int t = 0; t < 8; ++t) acc[t] = wmma_bf16(a, bf[t], acc[t]);
        wait_async();
        __syncthreads();
    }
    int h = lane >> 4, n = lane & 15;
#pragma unroll
    for (int t = 0; t < 8; ++t)
#pragma unroll
        for (int v = 0; v < 8; ++v) {
            int row = q0 + wave * 16 + h * 8 + v;
            int col = k0 + t * 16 + n;
            out[(size_t)row * Sc + col] = acc[t][v] + attn_bias[(size_t)row * Sc + col];
        }
}

// K3: in-place row softmax over S=2048 (one block per row, wave32 + LDS reduction)
__global__ __launch_bounds__(256) void softmax_rows(float* __restrict__ attn) {
    float* p = attn + (size_t)blockIdx.x * Sc;
    int tid = threadIdx.x;
    float4 a = ((const float4*)p)[tid];
    float4 b = ((const float4*)p)[tid + 256];
    float m = fmaxf(fmaxf(fmaxf(a.x, a.y), fmaxf(a.z, a.w)),
                    fmaxf(fmaxf(b.x, b.y), fmaxf(b.z, b.w)));
#pragma unroll
    for (int off = 16; off; off >>= 1) m = fmaxf(m, __shfl_xor(m, off, 32));
    __shared__ float redm[8], reds[8];
    if ((tid & 31) == 0) redm[tid >> 5] = m;
    __syncthreads();
    m = redm[0];
#pragma unroll
    for (int i = 1; i < 8; ++i) m = fmaxf(m, redm[i]);
    a.x = __expf(a.x - m); a.y = __expf(a.y - m); a.z = __expf(a.z - m); a.w = __expf(a.w - m);
    b.x = __expf(b.x - m); b.y = __expf(b.y - m); b.z = __expf(b.z - m); b.w = __expf(b.w - m);
    float s = (a.x + a.y) + (a.z + a.w) + (b.x + b.y) + (b.z + b.w);
#pragma unroll
    for (int off = 16; off; off >>= 1) s += __shfl_xor(s, off, 32);
    if ((tid & 31) == 0) reds[tid >> 5] = s;
    __syncthreads();
    s = 0.f;
#pragma unroll
    for (int i = 0; i < 8; ++i) s += reds[i];
    float inv = __frcp_rn(s);
    a.x *= inv; a.y *= inv; a.z *= inv; a.w *= inv;
    b.x *= inv; b.y *= inv; b.z *= inv; b.w *= inv;
    ((float4*)p)[tid] = a;
    ((float4*)p)[tid + 256] = b;
}

// K4: out = P.V  (P fp32 -> bf16 on VALU staging; V async-copied, pre-transposed bf16 [d][k])
__global__ __launch_bounds__(256) void pv_gemm(const float* __restrict__ attn,
                                               const unsigned short* __restrict__ vbT,
                                               float* __restrict__ outp) {
    __shared__ __align__(16) unsigned short ldsP[2][128 * LW];
    __shared__ __align__(16) unsigned short ldsV[2][128 * LW];
    int b = blockIdx.z, q0 = blockIdx.x * 128, d0 = blockIdx.y * 128;
    const float* P = attn + (size_t)b * Sc * Sc;
    const unsigned short* Vt = vbT + (size_t)b * Sc * Dc;  // [d][k]
    float* O = outp + (size_t)b * Sc * Dc;
    int tid = threadIdx.x, lane = tid & 31, wave = tid >> 5;
    v8f acc[8] = {};
    constexpr int NIT = Sc / 32;
    stage_f32<128, 256>(ldsP[0], P + (size_t)q0 * Sc, Sc, tid);
    stage_async<128, 256>(ldsV[0], Vt + (size_t)d0 * Sc, Sc, tid);
    wait_async();
    __syncthreads();
    for (int it = 0; it < NIT; ++it) {
        int cur = it & 1;
        if (it + 1 < NIT) {
            __builtin_prefetch(P + (size_t)q0 * Sc + (it + 1) * 32, 0, 1);
            stage_async<128, 256>(ldsV[cur ^ 1], Vt + (size_t)d0 * Sc + (it + 1) * 32, Sc, tid);
            stage_f32<128, 256>(ldsP[cur ^ 1], P + (size_t)q0 * Sc + (it + 1) * 32, Sc, tid);
        }
        v16bf a = load_frag_A(ldsP[cur], wave * 16 + (lane & 15), lane);
        v16bf bf[8];
#pragma unroll
        for (int t = 0; t < 8; ++t) bf[t] = load_frag_B(ldsV[cur], t * 16 + (lane & 15), lane);
#pragma unroll
        for (int t = 0; t < 8; ++t) acc[t] = wmma_bf16(a, bf[t], acc[t]);
        wait_async();
        __syncthreads();
    }
    int h = lane >> 4, n = lane & 15;
#pragma unroll
    for (int t = 0; t < 8; ++t)
#pragma unroll
        for (int v = 0; v < 8; ++v) {
            int row = q0 + wave * 16 + h * 8 + v;
            O[(size_t)row * Dc + d0 + t * 16 + n] = acc[t][v];
        }
}

extern "C" void kernel_launch(void* const* d_in, const int* in_sizes, int n_in,
                              void* d_out, int out_size, void* d_ws, size_t ws_size,
                              hipStream_t stream) {
    const float* query     = (const float*)d_in[0];
    const float* key       = (const float*)d_in[1];
    const float* value     = (const float*)d_in[2];
    const float* attn_bias = (const float*)d_in[3];
    const float* Wq        = (const float*)d_in[4];
    const float* bq        = (const float*)d_in[5];
    const float* Wk        = (const float*)d_in[6];
    const float* bk        = (const float*)d_in[7];

    float* out  = (float*)d_out;                       // [B,S,D]
    float* attn = out + (size_t)Bc * Sc * Dc;          // [B,S,S]

    unsigned short* qb  = (unsigned short*)d_ws;                 // bf16 [B,S,D]
    unsigned short* kb  = qb + (size_t)Bc * Sc * Dc;             // bf16 [B,S,D]
    unsigned short* vbT = kb + (size_t)Bc * Sc * Dc;             // bf16 [B,D,S]

    transpose_v<<<dim3(Sc / 64, Dc / 64, Bc), 256, 0, stream>>>(value, vbT);
    proj<<<dim3(Sc / 64, Dc / 64, 2 * Bc), 128, 0, stream>>>(query, key, Wq, bq, Wk, bk, qb, kb);
    scores<<<dim3(Sc / 128, Sc / 128, Bc), 256, 0, stream>>>(qb, kb, attn_bias, attn);
    softmax_rows<<<dim3(Bc * Sc), 256, 0, stream>>>(attn);
    pv_gemm<<<dim3(Sc / 128, Dc / 128, Bc), 256, 0, stream>>>(attn, vbT, out);
}
